// GlobalLocalAttention_20976620273938
// MI455X (gfx1250) — compile-verified, matrix operand order checked by reading, not run on hardware
//
#include <hip/hip_runtime.h>
#include <hip/hip_bf16.h>

typedef __attribute__((ext_vector_type(16))) _Float16 v16h;
typedef __attribute__((ext_vector_type(8)))  float    v8f;
typedef __attribute__((ext_vector_type(4)))  int      v4i;

#if __has_builtin(__builtin_amdgcn_global_load_async_to_lds_b128) && \
    __has_builtin(__builtin_amdgcn_s_wait_asynccnt)
#define HAVE_ASYNC_LDS 1
typedef __attribute__((address_space(1))) v4i* gv4i_ptr;   // global AS pointer to int4 vec
typedef __attribute__((address_space(3))) v4i* lv4i_ptr;   // LDS AS pointer to int4 vec
#endif

// Problem constants (from reference setup)
constexpr int B_  = 4;
constexpr int C_  = 128;
constexpr int H_  = 96;
constexpr int W_  = 96;
constexpr int HD  = 48;          // downsampled H (RATE=2)
constexpr int WD  = 48;          // downsampled W
constexpr int L_  = HD * WD;     // 2304
constexpr int KD  = C_ * 9;      // 1152 : im2col depth (c,dy,dx)
constexpr int MD  = C_ * 16;     // 2048 : deconv GEMM rows (c,ky,kx)

// ---------------------------------------------------------------- stage 1
__global__ void k_downsample(const float* __restrict__ f, const float* __restrict__ b,
                             float* __restrict__ f_ds, float* __restrict__ b_ds) {
  int idx = blockIdx.x * blockDim.x + threadIdx.x;
  int total = B_ * C_ * HD * WD;
  if (idx >= total) return;
  int x = idx % WD; int t = idx / WD;
  int y = t % HD;   t /= HD;
  int c = t % C_;   int bb = t / C_;
  size_t src = (((size_t)bb * C_ + c) * H_ + 2 * y) * W_ + 2 * x;
  f_ds[idx] = f[src];
  b_ds[idx] = b[src];
}

// mask patches (batch 0 only, per reference) : m[L][9], mm[L]
__global__ void k_mask(const float* __restrict__ mask, float* __restrict__ m,
                       float* __restrict__ mm) {
  int l = blockIdx.x * blockDim.x + threadIdx.x;
  if (l >= L_) return;
  int ly = l / WD, lx = l % WD;
  float sum = 0.f;
  for (int dy = 0; dy < 3; ++dy)
    for (int dx = 0; dx < 3; ++dx) {
      int yy = ly + dy - 1, xx = lx + dx - 1;
      float v = 0.f;
      if (yy >= 0 && yy < HD && xx >= 0 && xx < WD)
        v = mask[(size_t)(8 * yy) * (4 * W_) + 8 * xx];   // mask[0,0,::8,::8] patches
      m[l * 9 + dy * 3 + dx] = v;
      sum += v;
    }
  mm[l] = (sum == 0.f) ? 1.f : 0.f;
}

// ---------------------------------------------------------------- stage 2
// Per (b,l): q/k/v attention on 128x9 patches -> blended, L2-normalized conv
// kernel wn (f16, row-major [b][l][k=c*9+j]).  Small awkward GEMMs -> VALU;
// the tensor cores are reserved for the two big GEMMs.
__global__ void __launch_bounds__(288) k_attn_wn(
    const float* __restrict__ f_ds, const float* __restrict__ b_ds,
    const float* __restrict__ m,
    const float* __restrict__ wq, const float* __restrict__ bq,
    const float* __restrict__ wk, const float* __restrict__ bk,
    const float* __restrict__ wv, const float* __restrict__ bv,
    const float* __restrict__ beta2p,
    _Float16* __restrict__ wn_out) {
  __shared__ float s_fi[KD], s_wi[KD], s_v[KD];
  __shared__ float s_q[144], s_k[144], s_sim[81], s_attn[81];
  __shared__ float s_m[9];
  __shared__ float s_red[288];
  __shared__ float s_norm;

  const int l = blockIdx.x, bb = blockIdx.y, t = threadIdx.x;
  const int ly = l / WD, lx = l % WD;
  const float beta2 = beta2p[0];

  if (t < 9) s_m[t] = m[l * 9 + t];
  for (int r = 0; r < 4; ++r) {
    int idx = t + 288 * r;
    int c = idx / 9, j = idx % 9, dy = j / 3, dx = j % 3;
    int yy = ly + dy - 1, xx = lx + dx - 1;
    bool in = (yy >= 0 && yy < HD && xx >= 0 && xx < WD);
    size_t off = (((size_t)bb * C_ + c) * HD + yy) * WD + xx;
    s_fi[idx] = in ? f_ds[off] : 0.f;
    s_wi[idx] = in ? b_ds[off] : 0.f;
  }
  __syncthreads();

  if (t < 144) {                       // q = wq * fi_p + bq   (16x9)
    int d = t / 9, j = t % 9;
    float acc = bq[d];
    for (int c = 0; c < C_; ++c) acc += wq[d * C_ + c] * s_fi[c * 9 + j];
    s_q[t] = acc;
  } else {                             // k = wk * wi_p + bk   (16x9)
    int tt = t - 144; int d = tt / 9, j = tt % 9;
    float acc = bk[d];
    for (int c = 0; c < C_; ++c) acc += wk[d * C_ + c] * s_wi[c * 9 + j];
    s_k[tt] = acc;
  }
  __syncthreads();

  if (t < 81) {                        // sim[n][mc] = (q^T k)[n][mc] * m[mc]
    int n = t / 9, mc = t % 9;
    float acc = 0.f;
    for (int d = 0; d < 16; ++d) acc += s_q[d * 9 + n] * s_k[d * 9 + mc];
    s_sim[t] = acc * s_m[mc];
  }
  __syncthreads();

  if (t < 9) {                         // row softmax
    float mx = s_sim[t * 9];
    for (int mc = 1; mc < 9; ++mc) mx = fmaxf(mx, s_sim[t * 9 + mc]);
    float sm = 0.f;
    for (int mc = 0; mc < 9; ++mc) { float e = __expf(s_sim[t * 9 + mc] - mx); s_attn[t * 9 + mc] = e; sm += e; }
    float inv = 1.f / sm;
    for (int mc = 0; mc < 9; ++mc) s_attn[t * 9 + mc] *= inv;
  }
  __syncthreads();

  for (int r = 0; r < 4; ++r) {        // v = wv * fi_p + bv  (128x9)
    int idx = t + 288 * r;
    int c = idx / 9, j = idx % 9;
    float acc = bv[c];
    for (int c2 = 0; c2 < C_; ++c2) acc += wv[c * C_ + c2] * s_fi[c2 * 9 + j];
    s_v[idx] = acc;
  }
  __syncthreads();

  float ss = 0.f;
  float finv[4];
  for (int r = 0; r < 4; ++r) {        // final = v*attn^T, blend, accumulate |.|^2
    int idx = t + 288 * r;
    int c = idx / 9, j = idx % 9;
    float fsum = 0.f;
    for (int n = 0; n < 9; ++n) fsum += s_v[c * 9 + n] * s_attn[j * 9 + n];
    float mj = s_m[j];
    float fin = beta2 * s_fi[idx] * mj + (1.f - mj) * fsum;
    finv[r] = fin;
    ss += fin * fin;
  }
  s_red[t] = ss;
  __syncthreads();
  if (t == 0) {
    float tot = 0.f;
    for (int i = 0; i < 288; ++i) tot += s_red[i];
    s_norm = fmaxf(sqrtf(tot), 1e-4f);
  }
  __syncthreads();
  float invn = 1.f / s_norm;
  size_t base = ((size_t)bb * L_ + l) * KD;
  for (int r = 0; r < 4; ++r) {
    int idx = t + 288 * r;
    wn_out[base + idx] = (_Float16)(finv[r] * invn);
  }
}

// ---------------------------------------------------------------- stage 3
// im2col of f_ds, f16, stored K-major (transposed):  Xt[b][p=(py,px)][k=(c,dy,dx)]
__global__ void k_im2col(const float* __restrict__ f_ds, _Float16* __restrict__ Xt) {
  size_t idx = (size_t)blockIdx.x * blockDim.x + threadIdx.x;
  size_t total = (size_t)B_ * L_ * KD;
  if (idx >= total) return;
  int k = (int)(idx % KD); size_t t = idx / KD;
  int p = (int)(t % L_);   int bb = (int)(t / L_);
  int c = k / 9, j = k % 9, dy = j / 3, dx = j % 3;
  int py = p / WD, px = p % WD;
  int yy = py + dy - 1, xx = px + dx - 1;
  float v = 0.f;
  if (yy >= 0 && yy < HD && xx >= 0 && xx < WD)
    v = f_ds[(((size_t)bb * C_ + c) * HD + yy) * WD + xx];
  Xt[idx] = (_Float16)v;   // idx == ((bb*L_ + p)*KD + k)
}

// ---------------------------------------------------------------- WMMA GEMM
// C[MxN] = A[MxK] * Bt[NxK]^T.  A row-major f16 (lda=K), Bt is B stored
// K-major/transposed f16 (ld=K), C row-major f32.  One wave per 16x16 C tile,
// 4 waves per block sharing one N-tile; the shared 16x32 B tile is staged
// through double-buffered LDS with gfx1250 async-to-LDS copies (ASYNCcnt)
// when available.  Requires M%64==0, N%16==0, K%32==0.
// Per-lane register layouts per CDNA5 ISA 7.12.2:
//   A 16x32 f16 : lane<16 -> row=lane,    K in {0..7,16..23}
//                 lane>=16 -> row=lane-16, K in {8..15,24..31}
//   B 32x16 f16 : same K striping, N = lane&15
//   C 16x16 f32 : VGPR r -> row = r + (lane<16 ? 0 : 8), col = lane&15
__global__ void __launch_bounds__(128) k_wmma_gemm(
    const _Float16* __restrict__ A, const _Float16* __restrict__ Bt,
    float* __restrict__ C, int M, int N, int K) {
  __shared__ __align__(16) _Float16 sB[2][16 * 32];   // [buf][n][32 k-halves]
  const int lane  = threadIdx.x;
  const int wv    = threadIdx.y;
  const int tid   = wv * 32 + lane;
  const int tileN = blockIdx.x;
  const int tileM = blockIdx.y * 4 + wv;
  const int col     = lane & 15;
  const int halfSel = (lane < 16) ? 0 : 8;
  const _Float16* Arow = A + (size_t)(tileM * 16 + col) * K + halfSel;

  // staging assignment: 64 threads x 16B cover the 16x32-half B tile
  const int srow = tid >> 2, schunk = tid & 3;
  const _Float16* Bg = Bt + (size_t)(tileN * 16 + srow) * K + schunk * 8;

  auto stage = [&](int buf, int k0) {
    if (tid < 64) {
      _Float16* dst = &sB[buf][srow * 32 + schunk * 8];
#if HAVE_ASYNC_LDS
      // generic->AS1: identical 64-bit value; generic LDS ptr low 32 bits = LDS offset
      __builtin_amdgcn_global_load_async_to_lds_b128(
          (gv4i_ptr)(uintptr_t)(Bg + k0),
          (lv4i_ptr)(uint32_t)(uintptr_t)dst,
          0, 0);
#else
      *(uint4*)dst = *(const uint4*)(Bg + k0);
#endif
    }
  };

  v8f acc = {};
  stage(0, 0);
  for (int k0 = 0; k0 < K; k0 += 32) {
    int cur = (k0 >> 5) & 1;
#if HAVE_ASYNC_LDS
    __builtin_amdgcn_s_wait_asynccnt(0);
#endif
    __syncthreads();                    // sB[cur] ready for all waves
    if (k0 + 32 < K) stage(cur ^ 1, k0 + 32);
    __builtin_prefetch(Arow + k0 + 64, 0, 1);   // global_prefetch_b8 (A stream)
    v16h a, b;
    const _Float16* Bl = &sB[cur][col * 32 + halfSel];
#pragma unroll
    for (int h = 0; h < 16; ++h) {
      int off = (h < 8) ? h : (h + 8);
      a[h] = Arow[k0 + off];            // 2x global_load_b128
      b[h] = Bl[off];                   // 2x ds_load_b128
    }
    acc = __builtin_amdgcn_wmma_f32_16x16x32_f16(
        /*neg_a=*/false, a, /*neg_b=*/false, b,
        /*c_mod=*/(short)0, acc, /*reuse_a=*/false, /*reuse_b=*/false);
    __syncthreads();                    // reads of sB[cur] done before reuse
  }
  const int rowOff = (lane < 16) ? 0 : 8;
#pragma unroll
  for (int r = 0; r < 8; ++r)
    C[(size_t)(tileM * 16 + rowOff + r) * N + tileN * 16 + col] = acc[r];
}

// ---------------------------------------------------------------- stage 5
// pass 1 of eye-kernel smoothing: A[i,j] = sum_{s=-1..1} Y[i+s, j+s]
__global__ void k_smooth1(const float* __restrict__ Y, float* __restrict__ A) {
  size_t idx = (size_t)blockIdx.x * blockDim.x + threadIdx.x;
  if (idx >= (size_t)L_ * L_) return;
  int j = (int)(idx % L_), i = (int)(idx / L_);
  float acc = 0.f;
  for (int s = -1; s <= 1; ++s) {
    int ii = i + s, jj = j + s;
    if (ii >= 0 && ii < L_ && jj >= 0 && jj < L_) acc += Y[(size_t)ii * L_ + jj];
  }
  A[idx] = acc;
}

__device__ __forceinline__ int gmap(int i) { return (i % WD) * WD + (i / WD); }  // involution (HD==WD)

// pass 2 (transposed index space): Z[l,p] = sum_s A[g(g(l)+s), g(g(p)+s)]
__global__ void k_smooth2(const float* __restrict__ A, float* __restrict__ Z) {
  size_t idx = (size_t)blockIdx.x * blockDim.x + threadIdx.x;
  if (idx >= (size_t)L_ * L_) return;
  int p = (int)(idx % L_), l = (int)(idx / L_);
  int il = gmap(l), jp = gmap(p);
  float acc = 0.f;
  for (int s = -1; s <= 1; ++s) {
    int ii = il + s, jj = jp + s;
    if (ii >= 0 && ii < L_ && jj >= 0 && jj < L_)
      acc += A[(size_t)gmap(ii) * L_ + gmap(jj)];
  }
  Z[idx] = acc;
}

// ---------------------------------------------------------------- stage 6
// softmax over l (masked, SCALE=10); emits transposed f16 St[p][l] (GEMM2 Bt)
__global__ void __launch_bounds__(256) k_softmax(const float* __restrict__ Z,
                                                 const float* __restrict__ mm,
                                                 _Float16* __restrict__ St) {
  __shared__ float red[256];
  int p = blockIdx.x, t = threadIdx.x;
  float mx = -1e30f;
  for (int l = t; l < L_; l += 256)
    mx = fmaxf(mx, Z[(size_t)l * L_ + p] * mm[l] * 10.0f);
  red[t] = mx; __syncthreads();
  for (int s = 128; s > 0; s >>= 1) { if (t < s) red[t] = fmaxf(red[t], red[t + s]); __syncthreads(); }
  mx = red[0]; __syncthreads();
  float sm = 0.f;
  for (int l = t; l < L_; l += 256)
    sm += __expf(Z[(size_t)l * L_ + p] * mm[l] * 10.0f - mx);
  red[t] = sm; __syncthreads();
  for (int s = 128; s > 0; s >>= 1) { if (t < s) red[t] += red[t + s]; __syncthreads(); }
  float inv = 1.f / red[0];
  for (int l = t; l < L_; l += 256) {
    float v = Z[(size_t)l * L_ + p] * mm[l] * 10.0f;
    St[(size_t)p * L_ + l] = (_Float16)(__expf(v - mx) * inv * mm[l]);  // coalesced
  }
}

// Deconv A operand: Wd[(c,ky,kx)][l] = flipped raw_w patch of original b
__global__ void k_wdeconv(const float* __restrict__ bfull, int bb,
                          _Float16* __restrict__ Wd) {
  size_t idx = (size_t)blockIdx.x * blockDim.x + threadIdx.x;
  if (idx >= (size_t)MD * L_) return;
  int l = (int)(idx % L_); int mrow = (int)(idx / L_);
  int c = mrow / 16, kk = mrow % 16, ky = kk / 4, kx = kk % 4;
  int ly = l / WD, lx = l % WD;
  int ry = 2 * ly + 2 - ky, rx = 2 * lx + 2 - kx;   // raw_w[l,c,3-ky,3-kx]
  float v = 0.f;
  if (ry >= 0 && ry < H_ && rx >= 0 && rx < W_)
    v = bfull[(((size_t)bb * C_ + c) * H_ + ry) * W_ + rx];
  Wd[idx] = (_Float16)v;
}

// Gather the 4 valid (ky,kx) taps of the dilated conv into the output
__global__ void k_combine(const float* __restrict__ G, float* __restrict__ out) {
  size_t idx = (size_t)blockIdx.x * blockDim.x + threadIdx.x;
  if (idx >= (size_t)C_ * H_ * W_) return;
  int x = (int)(idx % W_); int t = (int)(idx / W_);
  int y = t % H_; int c = t / H_;
  float acc = 0.f;
  for (int ky = 0; ky < 4; ++ky) {
    int dy = y + ky - 2;
    if (dy < 0 || dy > 2 * HD - 2 || (dy & 1)) continue;
    int u = dy >> 1;
    for (int kx = 0; kx < 4; ++kx) {
      int dx = x + kx - 2;
      if (dx < 0 || dx > 2 * WD - 2 || (dx & 1)) continue;
      int v = dx >> 1;
      acc += G[(size_t)(c * 16 + ky * 4 + kx) * L_ + u * WD + v];
    }
  }
  out[idx] = acc * 0.25f;
}

// ---------------------------------------------------------------- launch
extern "C" void kernel_launch(void* const* d_in, const int* in_sizes, int n_in,
                              void* d_out, int out_size, void* d_ws, size_t ws_size,
                              hipStream_t stream) {
  const float* f     = (const float*)d_in[0];
  const float* b     = (const float*)d_in[1];
  const float* mask  = (const float*)d_in[2];
  const float* wq    = (const float*)d_in[3];
  const float* bq    = (const float*)d_in[4];
  const float* wk    = (const float*)d_in[5];
  const float* bk    = (const float*)d_in[6];
  const float* wv    = (const float*)d_in[7];
  const float* bv    = (const float*)d_in[8];
  const float* beta2 = (const float*)d_in[9];
  float* out = (float*)d_out;

  // carve workspace (~134 MB total)
  char* ws = (char*)d_ws;
  auto carve = [&](size_t bytes) { char* p = ws; ws += (bytes + 255) & ~(size_t)255; return p; };
  float*    f_ds = (float*)   carve((size_t)B_ * C_ * HD * WD * 4);
  float*    b_ds = (float*)   carve((size_t)B_ * C_ * HD * WD * 4);
  float*    mbuf = (float*)   carve((size_t)L_ * 9 * 4);
  float*    mm   = (float*)   carve((size_t)L_ * 4);
  _Float16* wn   = (_Float16*)carve((size_t)B_ * L_ * KD * 2);   // GEMM1 A
  _Float16* X16t = (_Float16*)carve((size_t)B_ * L_ * KD * 2);   // GEMM1 Bt [p][k]
  float*    Ybuf = (float*)   carve((size_t)L_ * L_ * 4);        // GEMM1 C / smooth2 out
  float*    Abuf = (float*)   carve((size_t)L_ * L_ * 4);        // smooth1 out
  _Float16* S16t = (_Float16*)carve((size_t)L_ * L_ * 2);        // GEMM2 Bt [p][l]
  _Float16* Wd16 = (_Float16*)carve((size_t)MD * L_ * 2);        // GEMM2 A
  float*    Gbuf = (float*)   carve((size_t)MD * L_ * 4);        // GEMM2 C

  const int TB = 256;
  auto cdiv = [](long long a, long long bb_) { return (int)((a + bb_ - 1) / bb_); };

  // global prep
  k_downsample<<<cdiv((long long)B_ * C_ * HD * WD, TB), TB, 0, stream>>>(f, b, f_ds, b_ds);
  k_mask<<<cdiv(L_, TB), TB, 0, stream>>>(mask, mbuf, mm);
  k_attn_wn<<<dim3(L_, B_), 288, 0, stream>>>(f_ds, b_ds, mbuf, wq, bq, wk, bk, wv, bv, beta2, wn);
  k_im2col<<<cdiv((long long)B_ * L_ * KD, TB), TB, 0, stream>>>(f_ds, X16t);

  // per-batch pipeline (sequential on stream -> L x L scratch reuse is safe)
  for (int bb = 0; bb < B_; ++bb) {
    // GEMM1: Y[l,p] = wn[bb] (2304x1152) * Xt[bb]^T (1152x2304)
    k_wmma_gemm<<<dim3(L_ / 16, L_ / 64), dim3(32, 4), 0, stream>>>(
        wn + (size_t)bb * L_ * KD, X16t + (size_t)bb * L_ * KD, Ybuf, L_, L_, KD);
    k_smooth1<<<cdiv((long long)L_ * L_, TB), TB, 0, stream>>>(Ybuf, Abuf);
    k_smooth2<<<cdiv((long long)L_ * L_, TB), TB, 0, stream>>>(Abuf, Ybuf);
    k_softmax<<<L_, 256, 0, stream>>>(Ybuf, mm, S16t);
    k_wdeconv<<<cdiv((long long)MD * L_, TB), TB, 0, stream>>>(b, bb, Wd16);
    // GEMM2: G = Wd (2048x2304) * St^T (2304x2304)
    k_wmma_gemm<<<dim3(L_ / 16, MD / 64), dim3(32, 4), 0, stream>>>(
        Wd16, S16t, Gbuf, MD, L_, L_);
    k_combine<<<cdiv((long long)C_ * H_ * W_, TB), TB, 0, stream>>>(
        Gbuf, out + (size_t)bb * C_ * H_ * W_);
  }
}
